// MultiHeadAttention_60833916780755
// MI455X (gfx1250) — matrix operand order repaired and checked by reading, not compiled, not run on hardware
//
#include <hip/hip_runtime.h>
#include <hip/hip_bf16.h>
#include <math.h>

typedef __attribute__((ext_vector_type(16))) _Float16 v16h;
typedef __attribute__((ext_vector_type(8)))  _Float16 v8h;
typedef __attribute__((ext_vector_type(8)))  float    v8f;

#define D_MODEL 768
#define NHEAD   12
#define DH      64
#define BATCH   2
#define SEQ     2048
#define MTOT    (BATCH*SEQ)

// ---- WMMA fragment helpers (wave32; contiguous K order k = 16*(lane/16) + s, same for A and B) ----

// A-matrix 16x32 f16: lane L holds row (L&15); halves 0..15 = K k0+(L>>4)*16..+15.
__device__ __forceinline__ v16h load_a_frag(const _Float16* base, int ld, int row0, int k0, int lane) {
  const int r  = row0 + (lane & 15);
  const int kA = k0 + ((lane >> 4) << 4);
  const _Float16* p = base + (size_t)r * ld + kA;
  v16h a;
  *(v8h*)&a       = *(const v8h*)(p);
  *((v8h*)&a + 1) = *(const v8h*)(p + 8);
  return a;
}

__device__ __forceinline__ v16h load_a_frag_lds(const _Float16* base, int ld, int lane) {
  const int r  = (lane & 15);
  const int kA = ((lane >> 4) << 4);
  const _Float16* p = base + r * ld + kA;
  v16h a;
  *(v8h*)&a       = *(const v8h*)(p);
  *((v8h*)&a + 1) = *(const v8h*)(p + 8);
  return a;
}

// B-matrix 32x16 f16 where memory holds B^T row-major ([N, K] rows contiguous in K):
// lane L = column col0+(L&15); 16 contiguous K at k0+(L>>4)*16.
__device__ __forceinline__ v16h load_b_frag(const _Float16* base, int ld, int col0, int k0, int lane) {
  const int n  = col0 + (lane & 15);
  const int ks = k0 + ((lane >> 4) << 4);
  return *(const v16h*)(base + (size_t)n * ld + ks);
}

__device__ __forceinline__ v8f wmma16(v16h a, v16h b, v8f c) {
  return __builtin_amdgcn_wmma_f32_16x16x32_f16(false, a, false, b, (short)0, c, false, false);
}

// ---- Kernel 1: f32 -> f16 conversion ----
__global__ void cvt_f32_to_f16(const float* __restrict__ src, _Float16* __restrict__ dst, int n) {
  int i = blockIdx.x * blockDim.x + threadIdx.x;
  if (i < n) dst[i] = (_Float16)src[i];
}

// ---- Kernel 2: fused QKV projection (y = x @ W^T + b), head-major outputs ----
// Per wave: 32x64 output tile (2 A-frags x 4 B-tiles -> 8 WMMAs per K-step).
// q,k stored [B,H,S,64]; v stored transposed [B,H,64,S] for the P*V GEMM.
__global__ void qkv_proj_kernel(const _Float16* __restrict__ Xh,
                                const _Float16* __restrict__ Wqh,
                                const _Float16* __restrict__ Wkh,
                                const _Float16* __restrict__ Wvh,
                                const float* __restrict__ bq,
                                const float* __restrict__ bk,
                                const float* __restrict__ bv,
                                _Float16* __restrict__ q16,
                                _Float16* __restrict__ k16,
                                _Float16* __restrict__ vT16) {
  const int lane = threadIdx.x & 31;
  const int wv   = threadIdx.x >> 5;
  const int m0   = (blockIdx.x * 4 + wv) * 32;
  const int h    = blockIdx.y;          // head index == 64-column group
  const int n0   = h * 64;
  const int sel  = blockIdx.z;
  const _Float16* W = (sel == 0) ? Wqh : (sel == 1) ? Wkh : Wvh;
  const float* bias = (sel == 0) ? bq  : (sel == 1) ? bk  : bv;

  v8f acc[8] = {};                      // acc[t*2+mt] : t = n-subtile, mt = m-subtile
  for (int k0 = 0; k0 < D_MODEL; k0 += 32) {
    v16h a0 = load_a_frag(Xh, D_MODEL, m0,      k0, lane);
    v16h a1 = load_a_frag(Xh, D_MODEL, m0 + 16, k0, lane);
    v16h b0 = load_b_frag(W, D_MODEL, n0 +  0, k0, lane);
    v16h b1 = load_b_frag(W, D_MODEL, n0 + 16, k0, lane);
    v16h b2 = load_b_frag(W, D_MODEL, n0 + 32, k0, lane);
    v16h b3 = load_b_frag(W, D_MODEL, n0 + 48, k0, lane);
    acc[0] = wmma16(a0, b0, acc[0]);  acc[1] = wmma16(a1, b0, acc[1]);
    acc[2] = wmma16(a0, b1, acc[2]);  acc[3] = wmma16(a1, b1, acc[3]);
    acc[4] = wmma16(a0, b2, acc[4]);  acc[5] = wmma16(a1, b2, acc[5]);
    acc[6] = wmma16(a0, b3, acc[6]);  acc[7] = wmma16(a1, b3, acc[7]);
  }

#pragma unroll
  for (int t = 0; t < 4; ++t) {
    const int d = t * 16 + (lane & 15);
    const float bval = bias[n0 + d];
#pragma unroll
    for (int mt = 0; mt < 2; ++mt) {
#pragma unroll
      for (int r = 0; r < 8; ++r) {
        const int m  = m0 + mt * 16 + r + ((lane >> 4) << 3);
        const int bb = m >> 11;           // SEQ == 2048
        const int sl = m & (SEQ - 1);
        const float val = acc[t * 2 + mt][r] + bval;
        if (sel == 0)
          q16[(((size_t)bb * NHEAD + h) * SEQ + sl) * DH + d] = (_Float16)val;
        else if (sel == 1)
          k16[(((size_t)bb * NHEAD + h) * SEQ + sl) * DH + d] = (_Float16)val;
        else
          vT16[(((size_t)bb * NHEAD + h) * DH + d) * SEQ + sl] = (_Float16)val;
      }
    }
  }
}

// ---- Kernel 3: flash attention, one wave per 16 query rows ----
__global__ void flash_attn_kernel(const _Float16* __restrict__ q16,
                                  const _Float16* __restrict__ k16,
                                  const _Float16* __restrict__ vT16,
                                  _Float16* __restrict__ O16) {
  __shared__ _Float16 Plds[8][16][32];      // per-wave P staging (C-layout -> A-layout)
  const int lane = threadIdx.x & 31;
  const int wv   = threadIdx.x >> 5;
  const int h    = blockIdx.y;
  const int bb   = blockIdx.z;
  const int i0   = (blockIdx.x * 8 + wv) * 16;

  const _Float16* qb = q16  + (((size_t)bb * NHEAD + h) * SEQ) * DH;
  const _Float16* kb = k16  + (((size_t)bb * NHEAD + h) * SEQ) * DH;
  const _Float16* vb = vT16 + (((size_t)bb * NHEAD + h) * DH) * SEQ;

  const v16h qA0 = load_a_frag(qb, DH, i0,  0, lane);
  const v16h qA1 = load_a_frag(qb, DH, i0, 32, lane);

  v8f o[4] = {};
  float mr[8], lr[8];
#pragma unroll
  for (int r = 0; r < 8; ++r) { mr[r] = -INFINITY; lr[r] = 0.f; }

  _Float16* P = &Plds[wv][0][0];
  const int prow = ((lane >> 4) << 3);
  const int pcol = (lane & 15);

  for (int jb = 0; jb < SEQ; jb += 32) {
    // prefetch next iteration's key/value fragments (global_prefetch_b8)
    if (jb + 32 < SEQ) {
      __builtin_prefetch(kb + (size_t)(jb + 32 + (lane & 15)) * DH + ((lane >> 4) << 4), 0, 1);
      __builtin_prefetch(vb + (size_t)(lane & 15) * SEQ + jb + 32 + ((lane >> 4) << 4), 0, 1);
    }

    // scores: two 16x16 tiles over keys [jb, jb+32)
    v8f c0 = {}, c1 = {};
    c0 = wmma16(qA0, load_b_frag(kb, DH, jb,       0, lane), c0);
    c0 = wmma16(qA1, load_b_frag(kb, DH, jb,      32, lane), c0);
    c1 = wmma16(qA0, load_b_frag(kb, DH, jb + 16,  0, lane), c1);
    c1 = wmma16(qA1, load_b_frag(kb, DH, jb + 16, 32, lane), c1);

    // online softmax; C-layout: row r+8*(lane>>4) lives in a 16-lane half
#pragma unroll
    for (int r = 0; r < 8; ++r) {
      float s0 = c0[r] * 0.125f;                 // 1/sqrt(64)
      float s1 = c1[r] * 0.125f;
      float t = fmaxf(s0, s1);
      t = fmaxf(t, __shfl_xor(t, 1));
      t = fmaxf(t, __shfl_xor(t, 2));
      t = fmaxf(t, __shfl_xor(t, 4));
      t = fmaxf(t, __shfl_xor(t, 8));
      const float mnew  = fmaxf(mr[r], t);
      const float alpha = __expf(mr[r] - mnew);
      mr[r] = mnew;
      const float p0 = __expf(s0 - mnew);
      const float p1 = __expf(s1 - mnew);
      float ps = p0 + p1;
      ps += __shfl_xor(ps, 1);
      ps += __shfl_xor(ps, 2);
      ps += __shfl_xor(ps, 4);
      ps += __shfl_xor(ps, 8);
      lr[r] = lr[r] * alpha + ps;
      o[0][r] *= alpha; o[1][r] *= alpha; o[2][r] *= alpha; o[3][r] *= alpha;
      P[(prow + r) * 32 + pcol]      = (_Float16)p0;
      P[(prow + r) * 32 + pcol + 16] = (_Float16)p1;
    }
    // P buffer is wave-private: LDS ops are in-order within a wave (DScnt),
    // so a zero-cost compiler scheduling fence is sufficient (no block barrier).
    __builtin_amdgcn_wave_barrier();

    const v16h pA = load_a_frag_lds(P, 32, lane);
#pragma unroll
    for (int t4 = 0; t4 < 4; ++t4)
      o[t4] = wmma16(pA, load_b_frag(vb, SEQ, t4 * 16, jb, lane), o[t4]);
    __builtin_amdgcn_wave_barrier();
  }

#pragma unroll
  for (int r = 0; r < 8; ++r) {
    const float inv = 1.0f / lr[r];
    const int m = bb * SEQ + i0 + r + ((lane >> 4) << 3);
#pragma unroll
    for (int t4 = 0; t4 < 4; ++t4) {
      const int n = h * 64 + t4 * 16 + (lane & 15);
      O16[(size_t)m * D_MODEL + n] = (_Float16)(o[t4][r] * inv);
    }
  }
}

// ---- Kernel 4: output projection + bias + residual (f32 out), 32x64 per wave ----
__global__ void out_proj_kernel(const _Float16* __restrict__ O16,
                                const _Float16* __restrict__ Woh,
                                const float* __restrict__ bo,
                                const float* __restrict__ resid,
                                float* __restrict__ xres) {
  const int lane = threadIdx.x & 31;
  const int wv   = threadIdx.x >> 5;
  const int m0   = (blockIdx.x * 4 + wv) * 32;
  const int n0   = blockIdx.y * 64;

  v8f acc[8] = {};
  for (int k0 = 0; k0 < D_MODEL; k0 += 32) {
    v16h a0 = load_a_frag(O16, D_MODEL, m0,      k0, lane);
    v16h a1 = load_a_frag(O16, D_MODEL, m0 + 16, k0, lane);
#pragma unroll
    for (int t = 0; t < 4; ++t) {
      v16h b = load_b_frag(Woh, D_MODEL, n0 + t * 16, k0, lane);
      acc[t * 2 + 0] = wmma16(a0, b, acc[t * 2 + 0]);
      acc[t * 2 + 1] = wmma16(a1, b, acc[t * 2 + 1]);
    }
  }
#pragma unroll
  for (int t = 0; t < 4; ++t) {
    const int n = n0 + t * 16 + (lane & 15);
    const float bval = bo[n];
#pragma unroll
    for (int mt = 0; mt < 2; ++mt) {
#pragma unroll
      for (int r = 0; r < 8; ++r) {
        const int m = m0 + mt * 16 + r + ((lane >> 4) << 3);
        xres[(size_t)m * D_MODEL + n] = acc[t * 2 + mt][r] + bval + resid[(size_t)m * D_MODEL + n];
      }
    }
  }
}

// ---- Kernel 5: LayerNorm over last dim (768 = 3*256) ----
__global__ void layernorm_kernel(const float* __restrict__ x,
                                 const float* __restrict__ gamma,
                                 const float* __restrict__ beta,
                                 float* __restrict__ out) {
  __shared__ float ssum[256];
  __shared__ float ssq[256];
  const int row = blockIdx.x;
  const int tid = threadIdx.x;
  const float* xr = x + (size_t)row * D_MODEL;
  const float v0 = xr[tid], v1 = xr[tid + 256], v2 = xr[tid + 512];
  ssum[tid] = v0 + v1 + v2;
  ssq[tid]  = v0 * v0 + v1 * v1 + v2 * v2;
  __syncthreads();
  for (int st = 128; st > 0; st >>= 1) {
    if (tid < st) { ssum[tid] += ssum[tid + st]; ssq[tid] += ssq[tid + st]; }
    __syncthreads();
  }
  const float mean = ssum[0] * (1.0f / D_MODEL);
  const float var  = ssq[0] * (1.0f / D_MODEL) - mean * mean;
  const float rstd = rsqrtf(var + 1e-5f);
  float* orow = out + (size_t)row * D_MODEL;
  orow[tid]       = (v0 - mean) * rstd * gamma[tid]       + beta[tid];
  orow[tid + 256] = (v1 - mean) * rstd * gamma[tid + 256] + beta[tid + 256];
  orow[tid + 512] = (v2 - mean) * rstd * gamma[tid + 512] + beta[tid + 512];
}

extern "C" void kernel_launch(void* const* d_in, const int* in_sizes, int n_in,
                              void* d_out, int out_size, void* d_ws, size_t ws_size,
                              hipStream_t stream) {
  (void)in_sizes; (void)n_in; (void)out_size; (void)ws_size;
  const float* Q   = (const float*)d_in[0];
  const float* Wq  = (const float*)d_in[1];
  const float* bq  = (const float*)d_in[2];
  const float* Wk  = (const float*)d_in[3];
  const float* bk  = (const float*)d_in[4];
  const float* Wv  = (const float*)d_in[5];
  const float* bv  = (const float*)d_in[6];
  const float* Wo  = (const float*)d_in[7];
  const float* bo  = (const float*)d_in[8];
  const float* g   = (const float*)d_in[9];
  const float* bta = (const float*)d_in[10];

  char* ws = (char*)d_ws;
  _Float16* Xh   = (_Float16*)(ws + 0);         //  4096*768 f16
  _Float16* Wqh  = (_Float16*)(ws + 6291456);   //  768*768 f16
  _Float16* Wkh  = (_Float16*)(ws + 7471104);
  _Float16* Wvh  = (_Float16*)(ws + 8650752);
  _Float16* Woh  = (_Float16*)(ws + 9830400);
  _Float16* q16  = (_Float16*)(ws + 11010048);  // [B,H,S,64] f16
  _Float16* k16  = (_Float16*)(ws + 17301504);  // [B,H,S,64] f16
  _Float16* vT16 = (_Float16*)(ws + 23592960);  // [B,H,64,S] f16
  _Float16* O16  = (_Float16*)(ws + 29884416);  // [M,768]    f16
  float*    xres = (float*)   (ws + 36175872);  // [M,768]    f32

  const int nX = MTOT * D_MODEL;
  const int nW = D_MODEL * D_MODEL;
  cvt_f32_to_f16<<<dim3((nX + 255) / 256), dim3(256), 0, stream>>>(Q,  Xh,  nX);
  cvt_f32_to_f16<<<dim3((nW + 255) / 256), dim3(256), 0, stream>>>(Wq, Wqh, nW);
  cvt_f32_to_f16<<<dim3((nW + 255) / 256), dim3(256), 0, stream>>>(Wk, Wkh, nW);
  cvt_f32_to_f16<<<dim3((nW + 255) / 256), dim3(256), 0, stream>>>(Wv, Wvh, nW);
  cvt_f32_to_f16<<<dim3((nW + 255) / 256), dim3(256), 0, stream>>>(Wo, Woh, nW);

  qkv_proj_kernel<<<dim3(MTOT / 128, NHEAD, 3), dim3(128), 0, stream>>>(
      Xh, Wqh, Wkh, Wvh, bq, bk, bv, q16, k16, vT16);

  flash_attn_kernel<<<dim3(SEQ / 128, NHEAD, BATCH), dim3(256), 0, stream>>>(
      q16, k16, vT16, O16);

  out_proj_kernel<<<dim3(MTOT / 128, NHEAD, 1), dim3(128), 0, stream>>>(
      O16, Woh, bo, Q, xres);

  layernorm_kernel<<<dim3(MTOT), dim3(256), 0, stream>>>(xres, g, bta, (float*)d_out);
}
